// Decoder_19988777796177
// MI455X (gfx1250) — matrix operand order repaired and checked by reading, not compiled, not run on hardware
//
#include <hip/hip_runtime.h>
#include <hip/hip_bf16.h>
#include <math.h>

// ---------------- problem constants ----------------
constexpr int Bk     = 256;
constexpr int Lk     = 64;
constexpr int Hk     = 768;
constexpr int NHEADk = 4;
constexpr int DHk    = Hk / NHEADk;     // 192
constexpr int DFFk   = 512;
constexpr int SLOTk  = 128;
constexpr int INTENTk= 32;
constexpr int NLAYERSk = 4;
constexpr int LSQk   = Lk / 4;          // 16
constexpr int Mk     = Bk * Lk;         // 16384 rows

// ---------------- WMMA types ----------------
typedef __attribute__((ext_vector_type(16))) _Float16 v16h;
typedef __attribute__((ext_vector_type(8)))  float    v8f;

#define AS1 __attribute__((address_space(1)))
#define AS3 __attribute__((address_space(3)))

// async-copy payload type: matches builtin signature (int vector_size 16)
typedef int vec4i_t __attribute__((vector_size(16)));

// gfx1250 async global->LDS copy path (ASYNCcnt-tracked), if the toolchain
// exposes the builtins; otherwise fall back to reg-staged b128 copies.
#if defined(__has_builtin)
#if __has_builtin(__builtin_amdgcn_global_load_async_to_lds_b128) && \
    __has_builtin(__builtin_amdgcn_s_wait_asynccnt)
#define HAVE_ASYNC_LDS 1
#endif
#endif
#ifndef HAVE_ASYNC_LDS
#define HAVE_ASYNC_LDS 0
#endif

// =====================================================================
// f16 GEMM: C[M,N] = act(A[M,K] @ W[N,K]^T + bias[N])
// Block tile 128x64, 256 threads = 8 waves (4x2); each wave owns a 32x32
// subtile (4 f32 accumulators). K stepped by 64 -> 8 WMMAs per stage.
// Async path: double-buffered LDS fed by global_load_async_to_lds_b128,
// next stage issued right after the barrier, overlapping the WMMAs.
// =====================================================================
#define TM 128
#define TN 64
#define TK 64
#define LPAD 8   // halves -> 16B pad, row stride 144B keeps b128 alignment

__device__ __forceinline__ void tile_mma(
    const _Float16 (*la)[TK + LPAD],
    const _Float16 (*lb)[TK + LPAD],
    v8f (&acc)[2][2], int wm, int wn, int mrow, int kb_a, int kb_b)
{
    #pragma unroll
    for (int kc = 0; kc < 2; ++kc) {              // two 32-wide K chunks
        v16h af[2], bf[2];
        #pragma unroll
        for (int i = 0; i < 2; ++i) {
            const _Float16* arow = &la[wm * 32 + i * 16 + mrow][kc * 32];
            #pragma unroll
            for (int r = 0; r < 8; ++r) {         // contiguous K: k = 16*(lane/16) + s
                const int k = kb_a + 2 * r;
                af[i][2 * r]     = arow[k];
                af[i][2 * r + 1] = arow[k + 1];
            }
            const _Float16* brow = &lb[wn * 32 + i * 16 + mrow][kc * 32];
            #pragma unroll
            for (int r = 0; r < 8; ++r) {         // B(k,n) = W[n,k]
                bf[i][2 * r]     = brow[kb_b + 2 * r];
                bf[i][2 * r + 1] = brow[kb_b + 2 * r + 1];
            }
        }
        #pragma unroll
        for (int i = 0; i < 2; ++i)
            #pragma unroll
            for (int j = 0; j < 2; ++j)
                acc[i][j] = __builtin_amdgcn_wmma_f32_16x16x32_f16(
                    false, af[i], false, bf[j], (short)0, acc[i][j], false, false);
    }
}

__global__ __launch_bounds__(256) void gemm_f16_wmma_kernel(
    const _Float16* __restrict__ A,     // M x K (f16)
    const _Float16* __restrict__ W,     // N x K (f16, torch layout)
    const float*    __restrict__ bias,  // N
    float*          __restrict__ C32,   // M x N or null
    _Float16*       __restrict__ C16,   // M x N or null
    int M, int N, int K, int relu)
{
#if HAVE_ASYNC_LDS
    __shared__ _Float16 lA[2][TM][TK + LPAD];
    __shared__ _Float16 lB[2][TN][TK + LPAD];
#else
    __shared__ _Float16 lA[1][TM][TK + LPAD];
    __shared__ _Float16 lB[1][TN][TK + LPAD];
#endif

    const int tid  = threadIdx.x;
    const int wave = tid >> 5;
    const int lane = tid & 31;
    const int wm   = wave >> 1;          // 0..3 -> 32-row strip
    const int wn   = wave & 1;           // 0..1 -> 32-col half
    const int tile_m = blockIdx.y * TM;
    const int tile_n = blockIdx.x * TN;

    v8f acc[2][2] = {};

    const int mrow = lane & 15;
    const int kb_a = (lane >> 4) * 16;
    const int kb_b = (lane >> 4) * 16;

    // staging assignment: A = 1024 16B segs (4/thread), B = 512 (2/thread)
    const int ar0 = tid >> 3;            // base row, +32 per segment step
    const int ac0 = (tid & 7) * 8;       // column (halves)

#if HAVE_ASYNC_LDS
    // ---------------- async double-buffered pipeline ----------------
    {   // prologue: stage k=0 into buffer 0
        #pragma unroll
        for (int s = 0; s < 4; ++s) {
            const int r = ar0 + s * 32;
            __builtin_amdgcn_global_load_async_to_lds_b128(
                (AS1 vec4i_t*)&A[(size_t)(tile_m + r) * K + ac0],
                (AS3 vec4i_t*)&lA[0][r][ac0], 0, 0);
        }
        #pragma unroll
        for (int s = 0; s < 2; ++s) {
            const int r = ar0 + s * 32;
            __builtin_amdgcn_global_load_async_to_lds_b128(
                (AS1 vec4i_t*)&W[(size_t)(tile_n + r) * K + ac0],
                (AS3 vec4i_t*)&lB[0][r][ac0], 0, 0);
        }
    }
    int p = 0;
    for (int k0 = 0; k0 < K; k0 += TK, p ^= 1) {
        __builtin_amdgcn_s_wait_asynccnt(0);   // my stage-p copies landed
        __syncthreads();                        // everyone's landed; prev reads of p^1 done
        if (k0 + TK < K) {                      // stage k0+TK into the other buffer
            const int kn = k0 + TK;
            #pragma unroll
            for (int s = 0; s < 4; ++s) {
                const int r = ar0 + s * 32;
                __builtin_amdgcn_global_load_async_to_lds_b128(
                    (AS1 vec4i_t*)&A[(size_t)(tile_m + r) * K + kn + ac0],
                    (AS3 vec4i_t*)&lA[p ^ 1][r][ac0], 0, 0);
            }
            #pragma unroll
            for (int s = 0; s < 2; ++s) {
                const int r = ar0 + s * 32;
                __builtin_amdgcn_global_load_async_to_lds_b128(
                    (AS1 vec4i_t*)&W[(size_t)(tile_n + r) * K + kn + ac0],
                    (AS3 vec4i_t*)&lB[p ^ 1][r][ac0], 0, 0);
            }
        }
        tile_mma(lA[p], lB[p], acc, wm, wn, mrow, kb_a, kb_b);
    }
#else
    // ---------------- fallback: reg-staged, all loads before stores ----------------
    for (int k0 = 0; k0 < K; k0 += TK) {
        uint4 ra[4], rb[2];
        #pragma unroll
        for (int s = 0; s < 4; ++s) {
            const int r = ar0 + s * 32;
            ra[s] = *(const uint4*)&A[(size_t)(tile_m + r) * K + k0 + ac0];
        }
        #pragma unroll
        for (int s = 0; s < 2; ++s) {
            const int r = ar0 + s * 32;
            rb[s] = *(const uint4*)&W[(size_t)(tile_n + r) * K + k0 + ac0];
        }
        if (k0 + TK < K) {
            __builtin_prefetch(&A[(size_t)(tile_m + (tid >> 1)) * K + k0 + TK], 0, 1);
            __builtin_prefetch(&W[(size_t)(tile_n + (tid >> 2)) * K + k0 + TK], 0, 1);
        }
        __syncthreads();                       // prev-iter reads complete
        #pragma unroll
        for (int s = 0; s < 4; ++s) *(uint4*)&lA[0][ar0 + s * 32][ac0] = ra[s];
        #pragma unroll
        for (int s = 0; s < 2; ++s) *(uint4*)&lB[0][ar0 + s * 32][ac0] = rb[s];
        __syncthreads();
        tile_mma(lA[0], lB[0], acc, wm, wn, mrow, kb_a, kb_b);
    }
#endif

    // C/D layout: N = lane&15, M = reg + 8*(lane>=16)
    const int nlo = lane & 15;
    const int mhi = (lane >> 4) * 8;
    #pragma unroll
    for (int i = 0; i < 2; ++i) {
        #pragma unroll
        for (int r = 0; r < 8; ++r) {
            const int gm = tile_m + wm * 32 + i * 16 + mhi + r;
            #pragma unroll
            for (int j = 0; j < 2; ++j) {
                const int gn = tile_n + wn * 32 + j * 16 + nlo;
                float v = acc[i][j][r] + bias[gn];
                if (relu) v = fmaxf(v, 0.0f);
                if (C32) C32[(size_t)gm * N + gn] = v;
                if (C16) C16[(size_t)gm * N + gn] = (_Float16)v;
            }
        }
    }
}

// =====================================================================
// fp32 -> f16 converter (weights), 4 elems/thread
// =====================================================================
__global__ __launch_bounds__(256) void cvt_f16_kernel(
    const float* __restrict__ src, _Float16* __restrict__ dst, long n)
{
    const long i = ((long)blockIdx.x * 256 + threadIdx.x) * 4;
    if (i + 3 < n) {
        const float4 v = *(const float4*)&src[i];
        dst[i + 0] = (_Float16)v.x;
        dst[i + 1] = (_Float16)v.y;
        dst[i + 2] = (_Float16)v.z;
        dst[i + 3] = (_Float16)v.w;
    }
}

// =====================================================================
// Embedding * sqrt(H) + batch-indexed positional encoding (reference bug);
// writes fp32 x (residual stream) and f16 mirror (GEMM input)
// =====================================================================
__global__ __launch_bounds__(256) void embed_pos_kernel(
    const int* __restrict__ inp, const float* __restrict__ emb,
    float* __restrict__ x, _Float16* __restrict__ x16)
{
    const size_t idx = (size_t)blockIdx.x * 256 + threadIdx.x;
    if (idx >= (size_t)Bk * Lk * Hk) return;
    const int    hcol = (int)(idx % Hk);
    const size_t bl   = idx / Hk;
    const int    b    = (int)(bl / Lk);
    const int    tok  = inp[bl];
    const float  v    = emb[(size_t)tok * Hk + hcol] * sqrtf((float)Hk);
    const int    i2   = hcol >> 1;
    const float  dv   = __expf((float)(2 * i2) * (-__logf(10000.0f) / (float)Hk));
    const float  ang  = (float)b * dv;
    const float  pe   = (hcol & 1) ? __cosf(ang) : __sinf(ang);
    const float  o    = v + pe;
    x[idx]   = o;
    x16[idx] = (_Float16)o;
}

// =====================================================================
// Stream compaction -> f16 newtensor (only ever consumed by a GEMM)
// =====================================================================
__global__ __launch_bounds__(256) void compact_kernel(
    const float* __restrict__ eo, const int* __restrict__ mask,
    _Float16* __restrict__ nt)
{
    const int b = blockIdx.x;
    __shared__ int pos[Lk];
    if (threadIdx.x == 0) {
        int c = 0;
        for (int l = 0; l < Lk; ++l)
            pos[l] = mask[b * Lk + l] ? (c++) : -1;
    }
    __syncthreads();
    _Float16* ntb = nt + (size_t)b * Lk * Hk;
    const float* eob = eo + (size_t)b * Lk * Hk;
    for (int i = threadIdx.x; i < Lk * Hk; i += 256) ntb[i] = (_Float16)0.0f;
    __syncthreads();
    for (int l = 0; l < Lk; ++l) {
        const int p = pos[l];
        if (p >= 0)
            for (int h = threadIdx.x; h < Hk; h += 256)
                ntb[p * Hk + h] = (_Float16)eob[l * Hk + h];
    }
}

// =====================================================================
// Attention scores + softmax per (b, head, q) row. mask_mode: 0=causal, 1=diag
// =====================================================================
__global__ __launch_bounds__(64) void attn_softmax_kernel(
    const float* __restrict__ Q, int qstride, int qoff,
    const float* __restrict__ KV, int kvstride, int koff,
    float* __restrict__ probs, int mask_mode)
{
    const int blk = blockIdx.x;                 // (b*NHEAD + h)*L + qi
    const int qi  = blk % Lk;
    const int h   = (blk / Lk) % NHEADk;
    const int b   = blk / (Lk * NHEADk);
    const int ki  = threadIdx.x;                // 0..63

    const float* qrow = Q  + (size_t)(b * Lk + qi) * qstride + qoff + h * DHk;
    const float* krow = KV + (size_t)(b * Lk + ki) * kvstride + koff + h * DHk;
    float s = 0.0f;
    for (int d = 0; d < DHk; ++d) s += qrow[d] * krow[d];
    s *= rsqrtf((float)DHk);
    const bool ok = (mask_mode == 0) ? (ki <= qi) : (ki == qi);
    s = ok ? s : -1e30f;

    __shared__ float sh[Lk];
    sh[ki] = s;
    __syncthreads();
    float mx = -1e30f;
    for (int j = 0; j < Lk; ++j) mx = fmaxf(mx, sh[j]);
    const float e = __expf(s - mx);
    __syncthreads();
    sh[ki] = e;
    __syncthreads();
    float sum = 0.0f;
    for (int j = 0; j < Lk; ++j) sum += sh[j];
    probs[(size_t)blk * Lk + ki] = e / sum;
}

// =====================================================================
// O[b,q,:] = probs @ V   (f16 output: feeds the out-projection GEMM)
// =====================================================================
__global__ __launch_bounds__(256) void attn_out_kernel(
    const float* __restrict__ probs,
    const float* __restrict__ KV, int kvstride, int voff,
    _Float16* __restrict__ O)
{
    const size_t idx = (size_t)blockIdx.x * 256 + threadIdx.x;
    if (idx >= (size_t)Bk * Lk * Hk) return;
    const int hd = (int)(idx % Hk);
    const int qi = (int)((idx / Hk) % Lk);
    const int b  = (int)(idx / ((size_t)Hk * Lk));
    const int h  = hd / DHk;
    const float* p = probs + (size_t)((b * NHEADk + h) * Lk + qi) * Lk;
    float acc = 0.0f;
    for (int k = 0; k < Lk; ++k)
        acc += p[k] * KV[(size_t)(b * Lk + k) * kvstride + voff + hd];
    O[idx] = (_Float16)acc;
}

// =====================================================================
// x = LayerNorm(x + d) * g + b; also refresh f16 mirror of x
// =====================================================================
__global__ __launch_bounds__(256) void add_ln_kernel(
    float* __restrict__ X, const float* __restrict__ D,
    const float* __restrict__ g, const float* __restrict__ be,
    _Float16* __restrict__ X16)
{
    const int row = blockIdx.x;
    float* x = X + (size_t)row * Hk;
    _Float16* x16 = X16 + (size_t)row * Hk;
    const float* d = D + (size_t)row * Hk;
    __shared__ float red[256];

    float vals[3];
    float local = 0.0f;
    #pragma unroll
    for (int i = 0; i < 3; ++i) {
        vals[i] = x[threadIdx.x + 256 * i] + d[threadIdx.x + 256 * i];
        local += vals[i];
    }
    red[threadIdx.x] = local;
    __syncthreads();
    for (int s = 128; s > 0; s >>= 1) {
        if ((int)threadIdx.x < s) red[threadIdx.x] += red[threadIdx.x + s];
        __syncthreads();
    }
    const float mu = red[0] * (1.0f / Hk);
    __syncthreads();

    local = 0.0f;
    #pragma unroll
    for (int i = 0; i < 3; ++i) { const float t = vals[i] - mu; local += t * t; }
    red[threadIdx.x] = local;
    __syncthreads();
    for (int s = 128; s > 0; s >>= 1) {
        if ((int)threadIdx.x < s) red[threadIdx.x] += red[threadIdx.x + s];
        __syncthreads();
    }
    const float rinv = rsqrtf(red[0] * (1.0f / Hk) + 1e-5f);
    __syncthreads();

    #pragma unroll
    for (int i = 0; i < 3; ++i) {
        const int col = threadIdx.x + 256 * i;
        const float o = (vals[i] - mu) * rinv * g[col] + be[col];
        x[col]   = o;
        x16[col] = (_Float16)o;
    }
}

// =====================================================================
// Intent branch
// =====================================================================
__global__ __launch_bounds__(256) void intent_squeeze_kernel(
    const float* __restrict__ eo, const float* __restrict__ sq_w,
    const float* __restrict__ sq_b, float* __restrict__ m)
{
    const int idx = blockIdx.x * 256 + threadIdx.x;
    if (idx >= Bk * Hk) return;
    const int h = idx % Hk;
    const int b = idx / Hk;
    float acc[LSQk];
    #pragma unroll
    for (int j = 0; j < LSQk; ++j) acc[j] = sq_b[j];
    for (int l = 0; l < Lk; ++l) {
        const float v = eo[((size_t)b * Lk + l) * Hk + h];
        #pragma unroll
        for (int j = 0; j < LSQk; ++j) acc[j] += v * sq_w[j * Lk + l];
    }
    float s = 0.0f;
    #pragma unroll
    for (int j = 0; j < LSQk; ++j) s += fmaxf(acc[j], 0.0f);
    m[idx] = s * (1.0f / LSQk);
}

__global__ __launch_bounds__(32) void intent_logits_kernel(
    const float* __restrict__ m, const float* __restrict__ int_w,
    const float* __restrict__ int_b, float* __restrict__ out)
{
    const int b = blockIdx.x;
    const int i = threadIdx.x;                    // 0..31
    const float* mb = m + (size_t)b * Hk;
    float acc = int_b[i];
    for (int hh = 0; hh < Hk; ++hh) acc += mb[hh] * int_w[i * Hk + hh];
    __shared__ float sh[INTENTk];
    sh[i] = acc;
    __syncthreads();
    float mx = -1e30f;
    for (int j = 0; j < INTENTk; ++j) mx = fmaxf(mx, sh[j]);
    float sum = 0.0f;
    for (int j = 0; j < INTENTk; ++j) sum += __expf(sh[j] - mx);
    out[(size_t)b * INTENTk + i] = (acc - mx) - __logf(sum);
}

// =====================================================================
// In-place log-softmax over SLOT=128 per row
// =====================================================================
__global__ __launch_bounds__(128) void slot_logsoftmax_kernel(float* __restrict__ S)
{
    const int row = blockIdx.x;
    const int i   = threadIdx.x;
    const float v = S[(size_t)row * SLOTk + i];
    __shared__ float sh[SLOTk];
    sh[i] = v;
    __syncthreads();
    float mx = -1e30f;
    for (int j = 0; j < SLOTk; ++j) mx = fmaxf(mx, sh[j]);
    float sum = 0.0f;
    for (int j = 0; j < SLOTk; ++j) sum += __expf(sh[j] - mx);
    S[(size_t)row * SLOTk + i] = (v - mx) - __logf(sum);
}

// =====================================================================
// host side
// =====================================================================
static inline void launch_gemm(const _Float16* A, const _Float16* W,
                               const float* bias, float* C32, _Float16* C16,
                               int M, int N, int K, int relu, hipStream_t s)
{
    dim3 grid(N / TN, M / TM), block(256);
    gemm_f16_wmma_kernel<<<grid, block, 0, s>>>(A, W, bias, C32, C16, M, N, K, relu);
}

static inline void launch_cvt(const float* src, _Float16* dst, long n, hipStream_t s)
{
    const long thr = (n + 3) / 4;
    cvt_f16_kernel<<<(int)((thr + 255) / 256), 256, 0, s>>>(src, dst, n);
}

extern "C" void kernel_launch(void* const* d_in, const int* in_sizes, int n_in,
                              void* d_out, int out_size, void* d_ws, size_t ws_size,
                              hipStream_t stream) {
    (void)in_sizes; (void)n_in; (void)out_size; (void)ws_size;

    const int*   inp     = (const int*)d_in[0];
    const float* eo      = (const float*)d_in[1];
    /* d_in[2] encoder_maskings: unused by reference math */
    const int*   sub     = (const int*)d_in[3];
    const float* emb     = (const float*)d_in[4];
    const float* sq_w    = (const float*)d_in[5];
    const float* sq_b    = (const float*)d_in[6];
    const float* int_w   = (const float*)d_in[7];
    const float* int_b   = (const float*)d_in[8];
    const float* slot_w  = (const float*)d_in[9];
    const float* slot_b  = (const float*)d_in[10];
    const float* sa_in_w = (const float*)d_in[11];
    const float* sa_in_b = (const float*)d_in[12];
    const float* sa_out_w= (const float*)d_in[13];
    const float* sa_out_b= (const float*)d_in[14];
    const float* ca_in_w = (const float*)d_in[15];
    const float* ca_in_b = (const float*)d_in[16];
    const float* ca_out_w= (const float*)d_in[17];
    const float* ca_out_b= (const float*)d_in[18];
    const float* ff1_w   = (const float*)d_in[19];
    const float* ff1_b   = (const float*)d_in[20];
    const float* ff2_w   = (const float*)d_in[21];
    const float* ff2_b   = (const float*)d_in[22];
    const float* ln1_g   = (const float*)d_in[23];
    const float* ln1_b   = (const float*)d_in[24];
    const float* ln2_g   = (const float*)d_in[25];
    const float* ln2_b   = (const float*)d_in[26];
    const float* ln3_g   = (const float*)d_in[27];
    const float* ln3_b   = (const float*)d_in[28];

    float* out_slot   = (float*)d_out;                   // (B*L, SLOT)
    float* out_intent = out_slot + (size_t)Mk * SLOTk;   // (B, INTENT)

    // ---- workspace carving (256B-aligned slabs) ----
    char* base = (char*)d_ws;
    size_t off = 0;
    auto carve = [&](size_t bytes) -> void* {
        void* p = base + off;
        off = (off + bytes + 255) & ~(size_t)255;
        return p;
    };
    float*     x      = (float*)    carve((size_t)Mk * Hk * 4);
    float*     qkv    = (float*)    carve((size_t)Mk * 3 * Hk * 4);
    float*     kv     = (float*)    carve((size_t)Mk * 2 * Hk * 4);
    float*     tmp    = (float*)    carve((size_t)Mk * Hk * 4);
    float*     probs  = (float*)    carve((size_t)Bk * NHEADk * Lk * Lk * 4);
    float*     im     = (float*)    carve((size_t)Bk * Hk * 4);
    _Float16*  x16    = (_Float16*) carve((size_t)Mk * Hk * 2);
    _Float16*  nt16   = (_Float16*) carve((size_t)Mk * Hk * 2);
    _Float16*  att16  = (_Float16*) carve((size_t)Mk * Hk * 2);
    _Float16*  ffm16  = (_Float16*) carve((size_t)Mk * DFFk * 2);
    _Float16*  saw16  = (_Float16*) carve((size_t)NLAYERSk * 3 * Hk * Hk * 2);
    _Float16*  saow16 = (_Float16*) carve((size_t)NLAYERSk * Hk * Hk * 2);
    _Float16*  caw16  = (_Float16*) carve((size_t)NLAYERSk * 3 * Hk * Hk * 2);
    _Float16*  caow16 = (_Float16*) carve((size_t)NLAYERSk * Hk * Hk * 2);
    _Float16*  ff1w16 = (_Float16*) carve((size_t)NLAYERSk * DFFk * Hk * 2);
    _Float16*  ff2w16 = (_Float16*) carve((size_t)NLAYERSk * Hk * DFFk * 2);
    _Float16*  slw16  = (_Float16*) carve((size_t)SLOTk * Hk * 2);

    const size_t nelem = (size_t)Bk * Lk * Hk;
    const int    nblk  = (int)((nelem + 255) / 256);

    // ---- one-time f16 weight mirrors ----
    launch_cvt(sa_in_w,  saw16,  (long)NLAYERSk * 3 * Hk * Hk, stream);
    launch_cvt(sa_out_w, saow16, (long)NLAYERSk * Hk * Hk,     stream);
    launch_cvt(ca_in_w,  caw16,  (long)NLAYERSk * 3 * Hk * Hk, stream);
    launch_cvt(ca_out_w, caow16, (long)NLAYERSk * Hk * Hk,     stream);
    launch_cvt(ff1_w,    ff1w16, (long)NLAYERSk * DFFk * Hk,   stream);
    launch_cvt(ff2_w,    ff2w16, (long)NLAYERSk * Hk * DFFk,   stream);
    launch_cvt(slot_w,   slw16,  (long)SLOTk * Hk,             stream);

    // ---- intent branch ----
    intent_squeeze_kernel<<<(Bk * Hk + 255) / 256, 256, 0, stream>>>(eo, sq_w, sq_b, im);
    intent_logits_kernel<<<Bk, 32, 0, stream>>>(im, int_w, int_b, out_intent);

    // ---- compaction + embedding ----
    compact_kernel<<<Bk, 256, 0, stream>>>(eo, sub, nt16);
    embed_pos_kernel<<<nblk, 256, 0, stream>>>(inp, emb, x, x16);

    // ---- decoder layers ----
    for (int l = 0; l < NLAYERSk; ++l) {
        const _Float16* saw  = saw16  + (size_t)l * 3 * Hk * Hk;
        const float*    sab  = sa_in_b  + (size_t)l * 3 * Hk;
        const _Float16* saow = saow16 + (size_t)l * Hk * Hk;
        const float*    saob = sa_out_b + (size_t)l * Hk;
        const _Float16* caw  = caw16  + (size_t)l * 3 * Hk * Hk;
        const float*    cab  = ca_in_b  + (size_t)l * 3 * Hk;
        const _Float16* caow = caow16 + (size_t)l * Hk * Hk;
        const float*    caob = ca_out_b + (size_t)l * Hk;

        // self-attention
        launch_gemm(x16, saw, sab, qkv, nullptr, Mk, 3 * Hk, Hk, 0, stream);
        attn_softmax_kernel<<<Bk * NHEADk * Lk, 64, 0, stream>>>(
            qkv, 3 * Hk, 0, qkv, 3 * Hk, Hk, probs, /*causal*/0);
        attn_out_kernel<<<nblk, 256, 0, stream>>>(probs, qkv, 3 * Hk, 2 * Hk, att16);
        launch_gemm(att16, saow, saob, tmp, nullptr, Mk, Hk, Hk, 0, stream);
        add_ln_kernel<<<Mk, 256, 0, stream>>>(x, tmp,
            ln1_g + (size_t)l * Hk, ln1_b + (size_t)l * Hk, x16);

        // cross-attention (Q from x, K/V from compacted newtensor)
        launch_gemm(x16, caw, cab, qkv, nullptr, Mk, Hk, Hk, 0, stream);       // Q only
        launch_gemm(nt16, caw + (size_t)Hk * Hk, cab + Hk, kv, nullptr,
                    Mk, 2 * Hk, Hk, 0, stream);                                // K,V
        attn_softmax_kernel<<<Bk * NHEADk * Lk, 64, 0, stream>>>(
            qkv, Hk, 0, kv, 2 * Hk, 0, probs, /*diag*/1);
        attn_out_kernel<<<nblk, 256, 0, stream>>>(probs, kv, 2 * Hk, Hk, att16);
        launch_gemm(att16, caow, caob, tmp, nullptr, Mk, Hk, Hk, 0, stream);
        add_ln_kernel<<<Mk, 256, 0, stream>>>(x, tmp,
            ln2_g + (size_t)l * Hk, ln2_b + (size_t)l * Hk, x16);

        // FFN (ff1 emits f16 directly for ff2)
        launch_gemm(x16, ff1w16 + (size_t)l * DFFk * Hk, ff1_b + (size_t)l * DFFk,
                    nullptr, ffm16, Mk, DFFk, Hk, /*relu*/1, stream);
        launch_gemm(ffm16, ff2w16 + (size_t)l * Hk * DFFk, ff2_b + (size_t)l * Hk,
                    tmp, nullptr, Mk, Hk, DFFk, 0, stream);
        add_ln_kernel<<<Mk, 256, 0, stream>>>(x, tmp,
            ln3_g + (size_t)l * Hk, ln3_b + (size_t)l * Hk, x16);
    }

    // ---- slot head ----
    launch_gemm(x16, slw16, slot_b, out_slot, nullptr, Mk, SLOTk, Hk, 0, stream);
    slot_logsoftmax_kernel<<<Mk, 128, 0, stream>>>(out_slot);
}
